// CSICell_74311524155829
// MI455X (gfx1250) — compile-verified, matrix operand order checked by reading, not run on hardware
//
#include <hip/hip_runtime.h>

// ---------------------------------------------------------------------------
// CSICell for MI455X (gfx1250, wave32, WMMA).
// Big GEMMs: bf16 WMMA 16x16x32, f32 accumulation. Weights pre-packed to
// bf16 N-major once (reused by all 64 M-tiles); activations pre-packed to
// bf16 so the GEMM inner loop is pure 16B bf16 loads -> LDS -> wmma.
// Argmax / cqi stage stays f32 VALU (discrete result).
// ---------------------------------------------------------------------------

typedef __attribute__((ext_vector_type(16))) __bf16 v16bf;
typedef __attribute__((ext_vector_type(8)))  __bf16 v8bf;
typedef __attribute__((ext_vector_type(8)))  float  v8f;
typedef __attribute__((ext_vector_type(4)))  float  v4f;

#define BATCH   8192
#define HID     1024
#define NCB     2048
#define NCAT    1040                  // 512 (W_q) + 16 (W_csi) + 512 (W_basis)
#define INV_S8  0.35355339059327373f  // 1/sqrt(8)

// ---------------------------------------------------------------------------
// Kernel A: UE feedback. One wave per batch row b. f32 VALU on purpose.
// ---------------------------------------------------------------------------
__global__ __launch_bounds__(256)
void ue_feedback(const float* __restrict__ H, const float* __restrict__ V,
                 float* __restrict__ cqi_out, float* __restrict__ m_out,
                 float* __restrict__ p_out, float* __restrict__ v_ws)
{
    const int wave = threadIdx.x >> 5;
    const int lane = threadIdx.x & 31;
    const int b = blockIdx.x * 8 + wave;

    float hr[8], hi[8];
#pragma unroll
    for (int p = 0; p < 8; ++p) {
        hr[p] = H[b * 64 + 2 * p]     * INV_S8;
        hi[p] = H[b * 64 + 2 * p + 1] * INV_S8;
    }

    float best = -1.0f;
    int   bestc = 0;
    const float2* V2 = (const float2*)V;   // (p, c) -> V2[p*2048 + c]
    for (int cb = 0; cb < NCB; cb += 32) {
        const int c = cb + lane;
        float sre = 0.f, sim = 0.f;
#pragma unroll
        for (int p = 0; p < 8; ++p) {
            const float2 vv = V2[p * NCB + c];
            sre += hr[p] * vv.x - hi[p] * vv.y;
            sim += hr[p] * vv.y + hi[p] * vv.x;
        }
        const float s = sre * sre + sim * sim;
        if (s > best) { best = s; bestc = c; }   // within-lane: first wins
    }
#pragma unroll
    for (int off = 16; off > 0; off >>= 1) {     // ties -> smaller index
        const float ob = __shfl_xor(best, off, 32);
        const int   oc = __shfl_xor(bestc, off, 32);
        if (ob > best || (ob == best && oc < bestc)) { best = ob; bestc = oc; }
    }
    const int m = bestc;

    float cre = 0.f, cim = 0.f, vr = 0.f, vi = 0.f;
    if (lane < 8) {
        const float2 vv = V2[lane * NCB + m];
        vr = vv.x; vi = vv.y;
        const float hrl = H[b * 64 + 2 * lane]     * INV_S8;
        const float hil = H[b * 64 + 2 * lane + 1] * INV_S8;
        cre = hrl * vr - hil * vi;
        cim = hrl * vi + hil * vr;
        v_ws[b * 16 + 2 * lane]     = vr;
        v_ws[b * 16 + 2 * lane + 1] = vi;
    }
    p_out[b * 64 + 2 * lane]     = (lane < 8) ? vr * INV_S8 : 0.f;
    p_out[b * 64 + 2 * lane + 1] = (lane < 8) ? vi * INV_S8 : 0.f;

#pragma unroll
    for (int off = 16; off > 0; off >>= 1) {
        cre += __shfl_xor(cre, off, 32);
        cim += __shfl_xor(cim, off, 32);
    }
    if (lane == 0) {
        cqi_out[b] = sqrtf(cre * cre + cim * cim);
        m_out[b]   = (float)m;
    }
}

// ---------------------------------------------------------------------------
// bias0[j] = b_h[j] + (1/sqrt(8)) * sum_{p<8} W_in[(p*8+p)*2][j]
// ---------------------------------------------------------------------------
__global__ void bias_prep(const float* __restrict__ W_in,
                          const float* __restrict__ b_h,
                          float* __restrict__ bias0)
{
    const int j = blockIdx.x * blockDim.x + threadIdx.x;
    if (j < HID) {
        float s = 0.f;
#pragma unroll
        for (int p = 0; p < 8; ++p) s += W_in[(p * 18) * HID + j];
        bias0[j] = b_h[j] + INV_S8 * s;
    }
}

// ---------------------------------------------------------------------------
// pack_bf16: flat f32 -> bf16, 8 elements per thread.
// ---------------------------------------------------------------------------
__global__ __launch_bounds__(256)
void pack_bf16(const float* __restrict__ src, __bf16* __restrict__ dst)
{
    const int i = blockIdx.x * 256 + threadIdx.x;   // v8 chunk index
    const v4f x0 = ((const v4f*)src)[2 * i];
    const v4f x1 = ((const v4f*)src)[2 * i + 1];
    v8bf o;
#pragma unroll
    for (int j = 0; j < 4; ++j) { o[j] = (__bf16)x0[j]; o[4 + j] = (__bf16)x1[j]; }
    ((v8bf*)dst)[i] = o;
}

// ---------------------------------------------------------------------------
// pack_wT: WT[n*1024 + k] = bf16(W[k][n]) via 64x64 LDS tile transpose.
// MODE 0: W_h (N=1024). MODE 1: concat [W_q|W_csi|W_basis] (N=1040).
// Reads coalesced along n, writes coalesced along k.
// ---------------------------------------------------------------------------
template <int MODE>
__global__ __launch_bounds__(256)
void pack_wT(const float* __restrict__ S0, const float* __restrict__ S1,
             const float* __restrict__ S2, __bf16* __restrict__ WT)
{
    __shared__ __bf16 t[64 * 72];
    const int k0 = blockIdx.x * 64;
    const int n0 = blockIdx.y * 64;
#pragma unroll
    for (int j = 0; j < 16; ++j) {
        const int idx = threadIdx.x + j * 256;
        const int r = idx >> 6;        // k-local
        const int c = idx & 63;        // n-local
        const int gk = k0 + r, gn = n0 + c;
        float w = 0.f;
        if (MODE == 0) {
            w = S0[(size_t)gk * 1024 + gn];
        } else {
            if      (gn < 512)  w = S0[(size_t)gk * 512 + gn];
            else if (gn < 528)  w = S1[(size_t)gk * 16  + (gn - 512)];
            else if (gn < NCAT) w = S2[(size_t)gk * 512 + (gn - 528)];
        }
        t[r * 72 + c] = (__bf16)w;
    }
    __syncthreads();
#pragma unroll
    for (int j = 0; j < 16; ++j) {
        const int idx = threadIdx.x + j * 256;
        const int rn = idx >> 6;       // n-local
        const int ck = idx & 63;       // k-local
        const int gn = n0 + rn;
        if (MODE == 0 || gn < NCAT)
            WT[(size_t)gn * 1024 + k0 + ck] = t[ck * 72 + rn];
    }
}

// ---------------------------------------------------------------------------
// WMMA GEMM: C(M x N) = Abf(M x 1024) * WT^T, all-bf16 streams, f32 acc.
// Block tile 128(M) x 64(N), K step 32; 8 waves in 4x2, each 2x2 WMMA tiles.
// Register-staged prefetch of the next K tile overlaps with the WMMAs.
// LDS rows padded to 40 bf16 (80B): conflict-free, 16B-aligned frag loads.
// Fragment layout per ISA 7.12.2 (16-bit operands):
//   lane L, elem e: row = L&15, K = (e>>3)*16 + (L>>4)*8 + (e&7),
// identical for A (row = M) and N-major B (row = N).
// MODE 0: out0 = tanh(acc + bias0[n] + cqi[m]*w512[n] + (mfl[m]/2048)*w513[n]),
//         also written as bf16 to hbf for GEMM2.
// MODE 1: cols [0,512)->out0 (Q_next), [512,528)->out1 (csiR), [528,1040)->out2 (Rr).
// ---------------------------------------------------------------------------
__device__ inline v16bf ldsFrag(const __bf16* rowBase, int half)
{
    const v8bf lo = *(const v8bf*)(rowBase + half * 8);
    const v8bf hi = *(const v8bf*)(rowBase + 16 + half * 8);
    return __builtin_shufflevector(lo, hi, 0, 1, 2, 3, 4, 5, 6, 7,
                                   8, 9, 10, 11, 12, 13, 14, 15);
}

template <int MODE>
__global__ __launch_bounds__(256)
void gemm_wmma(const __bf16* __restrict__ Abf,   // M x 1024, bf16 row-major
               const __bf16* __restrict__ WT,    // N x 1024, bf16 (N-major)
               const float* __restrict__ bias0,
               const float* __restrict__ w512,
               const float* __restrict__ w513,
               const float* __restrict__ cqi,
               const float* __restrict__ mfl,
               float* __restrict__ out0,
               float* __restrict__ out1,
               float* __restrict__ out2,
               __bf16* __restrict__ hbf)
{
    __shared__ __bf16 aSh[128 * 40];
    __shared__ __bf16 bSh[64 * 40];

    const int tid  = threadIdx.x;
    const int wave = tid >> 5;
    const int lane = tid & 31;
    const int half = lane >> 4;
    const int l15  = lane & 15;
    const int wm   = wave >> 1;          // 0..3 -> M quadrant
    const int wn   = wave & 1;           // 0..1 -> N half
    const int m0   = blockIdx.y * 128;
    const int n0   = blockIdx.x * 64;

    // per-thread staging slots: A rows (tid>>2) and (tid>>2)+64, B row tid>>2
    const int rA  = tid >> 2;            // 0..63
    const int c8  = (tid & 3) * 8;       // k sub-offset within 32
    const int gnB = n0 + rA;
    const __bf16* gA0 = Abf + (size_t)(m0 + rA) * 1024 + c8;
    const __bf16* gA1 = Abf + (size_t)(m0 + rA + 64) * 1024 + c8;
    const __bf16* gB  = WT  + (size_t)gnB * 1024 + c8;
    __bf16* sA0 = aSh + rA * 40 + c8;
    __bf16* sA1 = aSh + (rA + 64) * 40 + c8;
    __bf16* sB  = bSh + rA * 40 + c8;
    const bool bValid = (MODE == 0) || (gnB < NCAT);

    v8f c00 = {0.f,0.f,0.f,0.f,0.f,0.f,0.f,0.f};
    v8f c01 = c00, c10 = c00, c11 = c00;

    const __bf16* aRow0 = aSh + (wm * 32 +      l15) * 40;
    const __bf16* aRow1 = aSh + (wm * 32 + 16 + l15) * 40;
    const __bf16* bRow0 = bSh + (wn * 32 +      l15) * 40;
    const __bf16* bRow1 = bSh + (wn * 32 + 16 + l15) * 40;

    const v8bf zero8 = {};
    v8bf ra0 = *(const v8bf*)(gA0);
    v8bf ra1 = *(const v8bf*)(gA1);
    v8bf rb  = bValid ? *(const v8bf*)(gB) : zero8;

    for (int k0 = 0; k0 < HID; k0 += 32) {
        *(v8bf*)sA0 = ra0;
        *(v8bf*)sA1 = ra1;
        *(v8bf*)sB  = rb;
        __syncthreads();

        if (k0 + 32 < HID) {             // prefetch next K tile into registers
            ra0 = *(const v8bf*)(gA0 + k0 + 32);
            ra1 = *(const v8bf*)(gA1 + k0 + 32);
            rb  = bValid ? *(const v8bf*)(gB + k0 + 32) : zero8;
        }

        const v16bf a0 = ldsFrag(aRow0, half);
        const v16bf a1 = ldsFrag(aRow1, half);
        const v16bf b0 = ldsFrag(bRow0, half);
        const v16bf b1 = ldsFrag(bRow1, half);
        c00 = __builtin_amdgcn_wmma_f32_16x16x32_bf16(false, a0, false, b0, (short)0, c00, false, false);
        c01 = __builtin_amdgcn_wmma_f32_16x16x32_bf16(false, a0, false, b1, (short)0, c01, false, false);
        c10 = __builtin_amdgcn_wmma_f32_16x16x32_bf16(false, a1, false, b0, (short)0, c10, false, false);
        c11 = __builtin_amdgcn_wmma_f32_16x16x32_bf16(false, a1, false, b1, (short)0, c11, false, false);
        __syncthreads();
    }

    // epilogue: C/D layout -> M = r + 8*half (+16*mt), N = l15 (+16*nt)
#pragma unroll
    for (int mt = 0; mt < 2; ++mt) {
#pragma unroll
        for (int nt = 0; nt < 2; ++nt) {
            const v8f c = (mt == 0) ? (nt == 0 ? c00 : c01)
                                    : (nt == 0 ? c10 : c11);
            const int gn = n0 + wn * 32 + nt * 16 + l15;
            float bb = 0.f, wA = 0.f, wB = 0.f;
            if (MODE == 0) { bb = bias0[gn]; wA = w512[gn]; wB = w513[gn]; }
#pragma unroll
            for (int r = 0; r < 8; ++r) {
                const int gm = m0 + wm * 32 + mt * 16 + r + 8 * half;
                const float val = c[r];
                if (MODE == 0) {
                    const float hv = tanhf(val + bb + cqi[gm] * wA
                                           + (mfl[gm] * (1.0f / 2048.0f)) * wB);
                    out0[(size_t)gm * HID + gn] = hv;
                    hbf[(size_t)gm * HID + gn]  = (__bf16)hv;
                } else {
                    if      (gn < 512)  out0[(size_t)gm * 512 + gn] = val;
                    else if (gn < 528)  out1[(size_t)gm * 16 + (gn - 512)] = val;
                    else if (gn < NCAT) out2[(size_t)gm * 512 + (gn - 528)] = val;
                }
            }
        }
    }
}

// ---------------------------------------------------------------------------
// csi_next[b,n] = (R[b,n,:] . csic[b,:]) / ||R[b,n,:]||  (complex)
// ---------------------------------------------------------------------------
__global__ __launch_bounds__(256)
void csi_finalize(const float* __restrict__ csiR, const float* __restrict__ Rr,
                  float* __restrict__ csi_next)
{
    const int idx = blockIdx.x * 256 + threadIdx.x;   // b*32 + n
    const int b = idx >> 5;
    const int n = idx & 31;
    const float* R  = Rr   + (size_t)b * 512 + n * 16;
    const float* cs = csiR + (size_t)b * 16;
    float nrm = 0.f, are = 0.f, aim = 0.f;
#pragma unroll
    for (int k = 0; k < 8; ++k) {
        const float rre = R[2 * k], rim = R[2 * k + 1];
        const float cr  = cs[2 * k], ci = cs[2 * k + 1];
        nrm += rre * rre + rim * rim;
        are += rre * cr - rim * ci;
        aim += rre * ci + rim * cr;
    }
    const float inv = 1.0f / sqrtf(nrm);
    csi_next[(size_t)b * 64 + 2 * n]     = are * inv;
    csi_next[(size_t)b * 64 + 2 * n + 1] = aim * inv;
}

// ---------------------------------------------------------------------------
extern "C" void kernel_launch(void* const* d_in, const int* in_sizes, int n_in,
                              void* d_out, int out_size, void* d_ws, size_t ws_size,
                              hipStream_t stream)
{
    const float* H       = (const float*)d_in[0];
    const float* dec     = (const float*)d_in[1];
    const float* V       = (const float*)d_in[2];
    const float* W_in    = (const float*)d_in[3];
    const float* W_h     = (const float*)d_in[4];
    const float* b_h     = (const float*)d_in[5];
    const float* W_q     = (const float*)d_in[6];
    const float* W_csi   = (const float*)d_in[7];
    const float* W_basis = (const float*)d_in[8];

    float* out = (float*)d_out;
    const size_t OFF_Q   = 0;                       // (B,512)
    const size_t OFF_CQI = OFF_Q   + (size_t)BATCH * 512;
    const size_t OFF_H   = OFF_CQI + (size_t)BATCH;
    const size_t OFF_CSI = OFF_H   + (size_t)BATCH * HID;
    const size_t OFF_P   = OFF_CSI + (size_t)BATCH * 64;
    const size_t OFF_M   = OFF_P   + (size_t)BATCH * 64;
    float* q_next = out + OFF_Q;
    float* cqi    = out + OFF_CQI;
    float* hbuf   = out + OFF_H;
    float* csi_n  = out + OFF_CSI;
    float* pbuf   = out + OFF_P;
    float* mbuf   = out + OFF_M;

    // workspace: f32 regions first (all sizes multiples of 16B), then bf16
    float* ws    = (float*)d_ws;
    float* v_ws  = ws;                                   // B*16
    float* bias0 = v_ws  + (size_t)BATCH * 16;           // 1024
    float* csiR  = bias0 + HID;                          // B*16
    float* Rr    = csiR  + (size_t)BATCH * 16;           // B*512
    __bf16* Abf   = (__bf16*)(Rr + (size_t)BATCH * 512); // B*1024
    __bf16* Hbf   = Abf + (size_t)BATCH * HID;           // B*1024
    __bf16* WhT   = Hbf + (size_t)BATCH * HID;           // 1024*1024
    __bf16* WcatT = WhT + (size_t)HID * HID;             // 1040*1024

    // one-time packs (weights reused by 64 M-tile blocks; A reused by N-tiles)
    pack_bf16<<<(BATCH * HID / 8) / 256, 256, 0, stream>>>(dec, Abf);
    dim3 gw0(16, 16), gw1(16, 17);
    pack_wT<0><<<gw0, 256, 0, stream>>>(W_h, nullptr, nullptr, WhT);
    pack_wT<1><<<gw1, 256, 0, stream>>>(W_q, W_csi, W_basis, WcatT);

    ue_feedback<<<BATCH / 8, 256, 0, stream>>>(H, V, cqi, mbuf, pbuf, v_ws);
    bias_prep<<<4, 256, 0, stream>>>(W_in, b_h, bias0);

    // GEMM1: h = tanh(dec @ W_h + fused bias), 8192x1024x1024
    dim3 g1(HID / 64, BATCH / 128);
    gemm_wmma<0><<<g1, 256, 0, stream>>>(Abf, WhT,
                                         bias0, W_in + (size_t)512 * HID,
                                         W_in + (size_t)513 * HID, cqi, mbuf,
                                         hbuf, nullptr, nullptr, Hbf);

    // GEMM2: h @ [W_q | W_csi | W_basis], 8192x1024x1040
    dim3 g2((NCAT + 63) / 64, BATCH / 128);
    gemm_wmma<1><<<g2, 256, 0, stream>>>(Hbf, WcatT,
                                         nullptr, nullptr, nullptr, nullptr, nullptr,
                                         q_next, csiR, Rr, nullptr);

    csi_finalize<<<BATCH * 32 / 256, 256, 0, stream>>>(csiR, Rr, csi_n);
}
